// LlamaFastMLP_prior_59425167507992
// MI455X (gfx1250) — matrix operand order repaired and checked by reading, not compiled, not run on hardware
//
#include <hip/hip_runtime.h>
#include <hip/hip_bf16.h>

#define GROUP   128
#define DMODEL  4096
#define INTER   11008
#define SEQ     2048

typedef __attribute__((ext_vector_type(16))) _Float16 v16h;
typedef __attribute__((ext_vector_type(8)))  _Float16 v8h;
typedef __attribute__((ext_vector_type(4)))  _Float16 v4h;
typedef __attribute__((ext_vector_type(8)))  float    v8f;
typedef __attribute__((ext_vector_type(4)))  int      v4i;

static __device__ __forceinline__ v16h cat8(v8h lo, v8h hi) {
  return __builtin_shufflevector(lo, hi, 0,1,2,3,4,5,6,7,8,9,10,11,12,13,14,15);
}

// Low 32 bits of a generic pointer into LDS == wave-relative LDS byte offset
// (flat LDS aperture keeps the offset in addr[31:0] per CDNA5 ISA 10.2).
static __device__ __forceinline__ unsigned lds_off(const void* p) {
  return (unsigned)(unsigned long long)p;
}

// Async DMA of 16B global -> LDS, tracked on ASYNCcnt (not LOADcnt), so the
// WMMA loop's A-operand waits never block on weight streaming. NT hint keeps
// the once-streamed 540MB of weights from evicting L2-resident activations.
static __device__ __forceinline__ void async_ld_b128(unsigned lds_addr,
                                                     const void* gaddr) {
  asm volatile("global_load_async_to_lds_b128 %0, %1, off th:TH_LOAD_NT"
               :: "v"(lds_addr), "v"(gaddr) : "memory");
}
static __device__ __forceinline__ void wait_async0() {
  asm volatile("s_wait_asynccnt 0x0" ::: "memory");
}

// ---------------------------------------------------------------------------
// Kernel 0: x fp32 -> f16 (A operand), vectorized 4-wide.
// ---------------------------------------------------------------------------
__global__ void cvt_f32_to_f16(const float4* __restrict__ x,
                               v4h* __restrict__ y, int n4) {
  int i = blockIdx.x * blockDim.x + threadIdx.x;
  if (i < n4) {
    float4 f = x[i];
    v4h o = {(_Float16)f.x, (_Float16)f.y, (_Float16)f.z, (_Float16)f.w};
    y[i] = o;
  }
}

// ---------------------------------------------------------------------------
// Kernel 1: fused dequant(gate/up) GEMM + SwiGLU -> h (f16)
//   grid = (INTER/16, 2), block = 256 (8 waves)
//   Per chunk (K=256): async-stage raw int32 weights to LDS (ASYNCcnt)
//   for chunk c+1 *before* running the WMMA loop on chunk c, then packed-f16
//   dequant into the B tile. Weight HBM latency hides under WMMA.
// ---------------------------------------------------------------------------
#define KC1 256
#define NC1 (DMODEL / KC1)     // 16 chunks
#define NU1 8                  // (32 rows * 64 units) / 256 threads
#define PAD 8                  // 16B row pad: 16 rows -> 16 distinct bank groups

__global__ __launch_bounds__(256)
void mlp_gu_kernel(const _Float16* __restrict__ x16,
                   const int*   __restrict__ qw,
                   const int*   __restrict__ qz,
                   const float* __restrict__ sc,
                   _Float16* __restrict__ hbuf) {
  __shared__ int      lraw[32 * KC1];          // 32 KB raw int32 staging
  __shared__ _Float16 lw[32][KC1 + PAD];       // 17 KB dequantized B tile

  const int tid  = threadIdx.x;
  const int lane = tid & 31;
  const int wave = tid >> 5;
  const int n0   = blockIdx.x * 16;
  const int yoff = blockIdx.y * (SEQ / 2);
  const int r    = lane & 15;
  const int hi   = lane >> 4;

  // ---- async stage: each thread owns NU1 16B units; it alone re-reads them.
  auto stage = [&](int kc) {
#pragma unroll
    for (int j = 0; j < NU1; ++j) {
      const int u   = tid + j * 256;
      const int row = u >> 6;              // 64 units per row
      const int k4  = (u & 63) * 4;
      const int grow = (row < 16) ? (n0 + row) : (INTER + n0 + (row - 16));
      async_ld_b128(lds_off(&lraw[u * 4]),
                    qw + (long)grow * DMODEL + kc + k4);
    }
  };
  // ---- packed-f16 dequant: (q - z) * s  ->  v_pk_sub/mul_f16, ds_store_b64
  auto cvt = [&](int kc) {
#pragma unroll
    for (int j = 0; j < NU1; ++j) {
      const int u   = tid + j * 256;
      const int row = u >> 6;
      const int k4  = (u & 63) * 4;
      const int grow = (row < 16) ? (n0 + row) : (INTER + n0 + (row - 16));
      const int g   = (kc + k4) >> 7;      // group (same for 4 elems)
      const _Float16 zh = (_Float16)qz[grow * (DMODEL / GROUP) + g];
      const _Float16 sh = (_Float16)sc[grow * (DMODEL / GROUP) + g];
      const v4i q = *(const v4i*)&lraw[u * 4];
      const v4h qh = {(_Float16)q[0], (_Float16)q[1],
                      (_Float16)q[2], (_Float16)q[3]};
      const v4h zv = {zh, zh, zh, zh};
      const v4h sv = {sh, sh, sh, sh};
      *(v4h*)&lw[row][k4] = (qh - zv) * sv;
    }
  };

  v8f accg[8], accu[8];
#pragma unroll
  for (int t = 0; t < 8; ++t) {
    accg[t] = (v8f){0,0,0,0,0,0,0,0};
    accu[t] = (v8f){0,0,0,0,0,0,0,0};
  }

  // A addressing: uniform base + 32-bit byte offsets (saddr-friendly)
  const char* xb = (const char*)x16;
  const int mbase = yoff + wave * 128;
  unsigned rowoff[8];
#pragma unroll
  for (int t = 0; t < 8; ++t)
    rowoff[t] = (unsigned)(((mbase + t * 16 + r) * DMODEL + hi * 8) * 2);

  // prologue: stage + convert chunk 0
  stage(0);
  wait_async0();
  cvt(0);

  for (int c = 0; c < NC1; ++c) {
    __syncthreads();                       // lw tile ready for all waves
    const int kc = c * KC1;
    if (c + 1 < NC1) stage(kc + KC1);      // ASYNCcnt traffic under WMMAs

#pragma unroll 2
    for (int k = 0; k < KC1; k += 32) {
      const int kb = k + hi * 8;
      v16h Bg = cat8(*(const v8h*)&lw[r][kb],      *(const v8h*)&lw[r][kb + 16]);
      v16h Bu = cat8(*(const v8h*)&lw[16 + r][kb], *(const v8h*)&lw[16 + r][kb + 16]);
      const unsigned ko = (unsigned)(kc + k) * 2u;
#pragma unroll
      for (int t = 0; t < 8; ++t) {
        v16h A = cat8(*(const v8h*)(xb + rowoff[t] + ko),
                      *(const v8h*)(xb + rowoff[t] + ko + 32));
        accg[t] = __builtin_amdgcn_wmma_f32_16x16x32_f16(
            false, A, false, Bg, (short)0, accg[t], false, false);
        accu[t] = __builtin_amdgcn_wmma_f32_16x16x32_f16(
            false, A, false, Bu, (short)0, accu[t], false, false);
      }
    }

    if (c + 1 < NC1) {
      wait_async0();                       // my staged units have landed
      __syncthreads();                     // everyone done reading lw
      cvt(kc + KC1);
    }
  }

  // ---- fused SwiGLU epilogue: h = silu(gate) * up, stored f16
#pragma unroll
  for (int t = 0; t < 8; ++t) {
#pragma unroll
    for (int rr = 0; rr < 8; ++rr) {
      const float g  = accg[t][rr];
      const float u  = accu[t][rr];
      const float hv = g * (1.0f / (1.0f + __expf(-g))) * u;
      const int   m  = mbase + t * 16 + hi * 8 + rr;
      hbuf[(long)m * INTER + n0 + r] = (_Float16)hv;
    }
  }
}

// ---------------------------------------------------------------------------
// Kernel 2: fused dequant(down) GEMM: out = h @ W_down^T (fp32 out)
//   grid = (DMODEL/16, 2); K = 11008 = 43 chunks of 256
// ---------------------------------------------------------------------------
#define KC2 256
#define NC2 (INTER / KC2)      // 43
#define NU2 4                  // (16 rows * 64 units) / 256 threads

__global__ __launch_bounds__(256)
void mlp_down_kernel(const _Float16* __restrict__ hbuf,
                     const int*   __restrict__ qw,
                     const int*   __restrict__ qz,
                     const float* __restrict__ sc,
                     float* __restrict__ out) {
  __shared__ int      lraw[16 * KC2];          // 16 KB
  __shared__ _Float16 lw[16][KC2 + PAD];       // 8.4 KB

  const int tid  = threadIdx.x;
  const int lane = tid & 31;
  const int wave = tid >> 5;
  const int n0   = blockIdx.x * 16;
  const int yoff = blockIdx.y * (SEQ / 2);
  const int r    = lane & 15;
  const int hi   = lane >> 4;

  auto stage = [&](int kc) {
#pragma unroll
    for (int j = 0; j < NU2; ++j) {
      const int u   = tid + j * 256;
      const int row = u >> 6;
      const int k4  = (u & 63) * 4;
      async_ld_b128(lds_off(&lraw[u * 4]),
                    qw + (long)(n0 + row) * INTER + kc + k4);
    }
  };
  auto cvt = [&](int kc) {
#pragma unroll
    for (int j = 0; j < NU2; ++j) {
      const int u   = tid + j * 256;
      const int row = u >> 6;
      const int k4  = (u & 63) * 4;
      const int grow = n0 + row;
      const int g   = (kc + k4) >> 7;
      const _Float16 zh = (_Float16)qz[grow * (INTER / GROUP) + g];
      const _Float16 sh = (_Float16)sc[grow * (INTER / GROUP) + g];
      const v4i q = *(const v4i*)&lraw[u * 4];
      const v4h qh = {(_Float16)q[0], (_Float16)q[1],
                      (_Float16)q[2], (_Float16)q[3]};
      const v4h zv = {zh, zh, zh, zh};
      const v4h sv = {sh, sh, sh, sh};
      *(v4h*)&lw[row][k4] = (qh - zv) * sv;
    }
  };

  v8f acc[8];
#pragma unroll
  for (int t = 0; t < 8; ++t) acc[t] = (v8f){0,0,0,0,0,0,0,0};

  const char* hb = (const char*)hbuf;
  const int mbase = yoff + wave * 128;
  unsigned rowoff[8];
#pragma unroll
  for (int t = 0; t < 8; ++t)
    rowoff[t] = (unsigned)(((mbase + t * 16 + r) * INTER + hi * 8) * 2);

  stage(0);
  wait_async0();
  cvt(0);

  for (int c = 0; c < NC2; ++c) {
    __syncthreads();
    const int kc = c * KC2;
    if (c + 1 < NC2) stage(kc + KC2);

#pragma unroll 2
    for (int k = 0; k < KC2; k += 32) {
      const int kb = k + hi * 8;
      v16h B = cat8(*(const v8h*)&lw[r][kb], *(const v8h*)&lw[r][kb + 16]);
      const unsigned ko = (unsigned)(kc + k) * 2u;
#pragma unroll
      for (int t = 0; t < 8; ++t) {
        v16h A = cat8(*(const v8h*)(hb + rowoff[t] + ko),
                      *(const v8h*)(hb + rowoff[t] + ko + 32));
        acc[t] = __builtin_amdgcn_wmma_f32_16x16x32_f16(
            false, A, false, B, (short)0, acc[t], false, false);
      }
    }

    if (c + 1 < NC2) {
      wait_async0();
      __syncthreads();
      cvt(kc + KC2);
    }
  }

#pragma unroll
  for (int t = 0; t < 8; ++t) {
#pragma unroll
    for (int rr = 0; rr < 8; ++rr) {
      const int m = mbase + t * 16 + hi * 8 + rr;
      out[(long)m * DMODEL + n0 + r] = acc[t][rr];
    }
  }
}

// ---------------------------------------------------------------------------
extern "C" void kernel_launch(void* const* d_in, const int* in_sizes, int n_in,
                              void* d_out, int out_size, void* d_ws, size_t ws_size,
                              hipStream_t stream) {
  const float* x     = (const float*)d_in[0];
  const int*   gu_qw = (const int*)  d_in[1];
  const int*   gu_qz = (const int*)  d_in[2];
  const float* gu_sc = (const float*)d_in[3];
  const int*   dn_qw = (const int*)  d_in[4];
  const int*   dn_qz = (const int*)  d_in[5];
  const float* dn_sc = (const float*)d_in[6];
  float*       out   = (float*)d_out;

  // workspace: x16 (16 MB) | h (45 MB) -- both L2-resident (192 MB L2)
  _Float16* x16  = (_Float16*)d_ws;
  _Float16* hbuf = x16 + (size_t)SEQ * DMODEL;

  const int n4 = SEQ * DMODEL / 4;
  cvt_f32_to_f16<<<dim3((n4 + 255) / 256), dim3(256), 0, stream>>>(
      (const float4*)x, (v4h*)x16, n4);
  mlp_gu_kernel<<<dim3(INTER / 16, 2), dim3(256), 0, stream>>>(
      x16, gu_qw, gu_qz, gu_sc, hbuf);
  mlp_down_kernel<<<dim3(DMODEL / 16, 2), dim3(256), 0, stream>>>(
      hbuf, dn_qw, dn_qz, dn_sc, out);
}
